// Transformer_23055384445755
// MI455X (gfx1250) — matrix operand order repaired and checked by reading, not compiled, not run on hardware
//
#include <hip/hip_runtime.h>
#include <hip/hip_bf16.h>
#include <cmath>

// ---------------------------------------------------------------------------
// Types for CDNA5 WMMA (wave32): D(f32 16x16) = A(bf16 16x32) x B(bf16 32x16) + C
// ---------------------------------------------------------------------------
typedef __bf16 bf16_t;
typedef __attribute__((ext_vector_type(16))) __bf16 v16bf;
typedef __attribute__((ext_vector_type(8)))  __bf16 v8bf;
typedef __attribute__((ext_vector_type(8)))  float  v8f;

#define NNODES 50000
#define NEDGES 800000

// Monotone float->uint encoding so unsigned atomicMax implements float max.
__device__ __forceinline__ unsigned enc_f32(float x) {
  unsigned u = __float_as_uint(x);
  return (u & 0x80000000u) ? ~u : (u | 0x80000000u);
}
__device__ __forceinline__ float dec_f32(unsigned u) {
  unsigned b = (u & 0x80000000u) ? (u & 0x7fffffffu) : ~u;
  return __uint_as_float(b);
}
#define ENC_NEG_INF 0x007FFFFFu   // enc(-inf)

// ---------------------------------------------------------------------------
// Utility kernels
// ---------------------------------------------------------------------------
__global__ void k_fill_u32(unsigned* __restrict__ p, unsigned v, int n) {
  int t = blockIdx.x * blockDim.x + threadIdx.x;
  if (t < n) p[t] = v;
}

__global__ void k_cvt_bf16(const float* __restrict__ in, bf16_t* __restrict__ out, int n) {
  int t = blockIdx.x * blockDim.x + threadIdx.x;
  if (t < n) out[t] = (bf16_t)in[t];
}

// Pack W[din x dout] (row-major f32) into transposed padded bf16 Wt[segBase+segPad][din],
// and bias (padded with zeros).
__global__ void k_pack_w(const float* __restrict__ w, const float* __restrict__ b,
                         bf16_t* __restrict__ Wt, float* __restrict__ bias,
                         int din, int dout, int segBase, int segPad) {
  int t = blockIdx.x * blockDim.x + threadIdx.x;
  int total = segPad * din;
  if (t < total) {
    int c  = t / din;
    int kk = t - c * din;
    float v = (c < dout) ? w[(size_t)kk * dout + c] : 0.0f;
    Wt[(size_t)(segBase + c) * din + kk] = (bf16_t)v;
  }
  if (t < segPad) bias[segBase + t] = (t < dout) ? b[t] : 0.0f;
}

// ---------------------------------------------------------------------------
// Fused QKVS GEMM: C[N x strideC] = A[N x DIN](bf16) * Wt^T + bias, via WMMA.
// Templated on DIN and tile count NT -> fully unrolled, branch-free inner loop
// (no predicated WMMA => no duplicated accumulator register sets).
// One wave per 16-row tile and NT 16-column tiles (register blocked).
// A frag (16x32 bf16): lane m=lane&15; lanes 0-15 hold K 0..7 & 16..23,
// lanes 16-31 hold K 8..15 & 24..31 (two contiguous 16B loads).
// B frag (32x16 bf16): lane = column; lanes 0-15 hold K 0..7 & 16..23, lanes 16-31
// hold K 8..15 & 24..31 (contiguous along K thanks to the transposed Wt layout).
// ---------------------------------------------------------------------------
template <int DIN, int NT>
__global__ __launch_bounds__(32)
void gemm_qkvs_wmma(const bf16_t* __restrict__ A,
                    const bf16_t* __restrict__ Wt,
                    const float*  __restrict__ bias,
                    float* __restrict__ C,
                    int colBase, int strideC) {
  const int lane  = threadIdx.x;
  const int laneM = lane & 15;
  const int hi    = lane >> 4;
  const int rm    = blockIdx.x << 4;                   // row tile base
  const int cg    = colBase + blockIdx.y * (NT * 16);  // column group base

  v8f acc[NT] = {};

  const bf16_t* arow = A + (size_t)(rm + laneM) * DIN;

#pragma unroll
  for (int k0 = 0; k0 < DIN; k0 += 32) {
    union { v16bf v; v8bf h[2]; } au;
    au.h[0] = *(const v8bf*)(arow + k0 + hi * 8);
    au.h[1] = *(const v8bf*)(arow + k0 + 16 + hi * 8);
#pragma unroll
    for (int nt = 0; nt < NT; ++nt) {
      const bf16_t* bcol = Wt + (size_t)(cg + nt * 16 + laneM) * DIN + k0 + hi * 8;
      union { v16bf v; v8bf h[2]; } bu;
      bu.h[0] = *(const v8bf*)(bcol);
      bu.h[1] = *(const v8bf*)(bcol + 16);
      acc[nt] = __builtin_amdgcn_wmma_f32_16x16x32_bf16(
          false, au.v, false, bu.v, (short)0, acc[nt], false, false);
    }
  }

#pragma unroll
  for (int nt = 0; nt < NT; ++nt) {
    const int col  = cg + nt * 16 + laneM;
    const float bv = bias[col];
#pragma unroll
    for (int r = 0; r < 8; ++r) {
      // C/D layout: VGPR r -> row r (lanes 0-15) / row r+8 (lanes 16-31)
      C[(size_t)(rm + r + hi * 8) * strideC + col] = acc[nt][r] + bv;
    }
  }
}

// ---------------------------------------------------------------------------
// Edge phase kernels (all data L2-resident: ~160MB < 192MB L2)
// ---------------------------------------------------------------------------
__global__ void k_edge_logits(const int* __restrict__ src, const int* __restrict__ dst,
                              const float* __restrict__ proj, int stride, int qb, int kb,
                              int D, float scale,
                              float* __restrict__ alpha, unsigned* __restrict__ amax,
                              int EH) {
  int t = blockIdx.x * blockDim.x + threadIdx.x;
  if (t >= EH) return;
  int e = t >> 1, h = t & 1;                 // H == 2
  int s = src[e], d = dst[e];
  const float* qp = proj + (size_t)d * stride + qb + h * D;
  const float* kp = proj + (size_t)s * stride + kb + h * D;
  float a = 0.0f;
  for (int j = 0; j < D; ++j) a += qp[j] * kp[j];
  a *= scale;
  alpha[t] = a;
  atomicMax(&amax[((size_t)d << 1) + h], enc_f32(a));
}

__global__ void k_edge_exp(const int* __restrict__ dst,
                           float* __restrict__ alpha, const unsigned* __restrict__ amax,
                           float* __restrict__ denom, int EH) {
  int t = blockIdx.x * blockDim.x + threadIdx.x;
  if (t >= EH) return;
  int e = t >> 1, h = t & 1;
  int d = dst[e];
  float m  = dec_f32(amax[((size_t)d << 1) + h]);
  float ex = expf(alpha[t] - m);
  alpha[t] = ex;                             // overwrite in place
  atomicAdd(&denom[((size_t)d << 1) + h], ex);
}

__global__ void k_edge_scatter(const int* __restrict__ src, const int* __restrict__ dst,
                               const float* __restrict__ proj, int stride, int vb,
                               const float* __restrict__ exbuf, const float* __restrict__ denom,
                               float* __restrict__ agg, int D, int HD, int total) {
  int t = blockIdx.x * blockDim.x + threadIdx.x;
  if (t >= total) return;
  int e = t / HD;
  int j = t - e * HD;
  int h = j / D;
  int s = src[e], d = dst[e];
  float attn = exbuf[((size_t)e << 1) + h] / (denom[((size_t)d << 1) + h] + 1e-16f);
  float val  = proj[(size_t)s * stride + vb + j] * attn;
  atomicAdd(&agg[(size_t)d * HD + j], val);
}

// ---------------------------------------------------------------------------
// Finalize: concat / head-mean + skip (+ReLU)
// ---------------------------------------------------------------------------
__global__ void k_finalize_concat(const float* __restrict__ agg, const float* __restrict__ proj,
                                  int stride, int sb, float* __restrict__ out,
                                  int HD, int doRelu, int total) {
  int t = blockIdx.x * blockDim.x + threadIdx.x;
  if (t >= total) return;
  int n = t / HD;
  int c = t - n * HD;
  float v = agg[t] + proj[(size_t)n * stride + sb + c];   // bias already in proj
  out[t] = doRelu ? fmaxf(v, 0.0f) : v;
}

__global__ void k_finalize_mean(const float* __restrict__ agg, const float* __restrict__ proj,
                                int stride, int sb, float* __restrict__ out,
                                int D, int HD, int doRelu, int total) {
  int t = blockIdx.x * blockDim.x + threadIdx.x;
  if (t >= total) return;
  int n = t / D;
  int c = t - n * D;
  float v = 0.5f * (agg[(size_t)n * HD + c] + agg[(size_t)n * HD + D + c])
          + proj[(size_t)n * stride + sb + c];
  out[t] = doRelu ? fmaxf(v, 0.0f) : v;
}

__global__ void k_log_softmax(const float* __restrict__ in, float* __restrict__ out,
                              int C, int n) {
  int t = blockIdx.x * blockDim.x + threadIdx.x;
  if (t >= n) return;
  const float* r = in + (size_t)t * C;
  float m = -3.402823466e38f;
  for (int c = 0; c < C; ++c) m = fmaxf(m, r[c]);
  float s = 0.0f;
  for (int c = 0; c < C; ++c) s += expf(r[c] - m);
  float ls = logf(s);
  float* o = out + (size_t)t * C;
  for (int c = 0; c < C; ++c) o[c] = r[c] - m - ls;
}

// ---------------------------------------------------------------------------
// Host driver: 3 layers, all kernels on `stream` (graph-capture safe).
// Input flattening: x, edge_index, then per layer {q.w,q.b,k.w,k.b,v.w,v.b,s.w,s.b}.
// ---------------------------------------------------------------------------
extern "C" void kernel_launch(void* const* d_in, const int* in_sizes, int n_in,
                              void* d_out, int out_size, void* d_ws, size_t ws_size,
                              hipStream_t stream) {
  const float* x  = (const float*)d_in[0];
  const int*   ei = (const int*)d_in[1];
  const int* src = ei;
  const int* dst = ei + NEDGES;

  char* ws = (char*)d_ws;
  bf16_t*   h16    = (bf16_t*)(ws + 0);            // 12,800,000 B (N x 128 bf16)
  float*    proj   = (float*)(ws + 12800000);      // 89,600,000 B (N x 448 f32)
  float*    alpha  = (float*)(ws + 102400000);     //  6,400,000 B (E x 2 f32)
  unsigned* amax   = (unsigned*)(ws + 108800000);  //    400,000 B (N x 2 u32)
  float*    denomb = (float*)(ws + 109200000);     //    400,000 B
  float*    agg    = (float*)(ws + 109600000);     // 25,600,000 B (N x 128 f32)
  float*    h1     = (float*)(ws + 135200000);     // 12,800,000 B (N x 64)
  float*    h2     = (float*)(ws + 148000000);     // 12,800,000 B (N x 64)
  bf16_t*   Wt     = (bf16_t*)(ws + 160800000);    //    131,072 B (448 x 128 bf16 max)
  float*    biasb  = (float*)(ws + 160931072);     //      2,048 B

  const int H = 2;
  const int dinA[3]  = {128, 64, 64};
  const int Dh[3]    = {32, 64, 40};
  const int HDh[3]   = {64, 128, 80};
  const int tp[3]    = {256, 448, 288};            // fused padded column counts
  const int qb[3]    = {0, 0, 0};
  const int kb[3]    = {64, 128, 80};
  const int vb[3]    = {128, 256, 160};
  const int sb[3]    = {192, 384, 240};
  const int sPad[3]  = {64, 64, 48};               // skip segment pad
  const int sOut[3]  = {64, 64, 40};               // skip actual dout
  const int concat[3] = {1, 0, 0};
  const int relu[3]   = {1, 1, 0};
  const float* lin[3] = {x, h1, h2};
  float* lout[3]      = {h1, h2, h1};              // layer3 pre-softmax reuses h1

  const int ROWT = NNODES / 16;                    // 3125 row tiles (exact)

  for (int L = 0; L < 3; ++L) {
    const int din = dinA[L], D = Dh[L], HD = HDh[L], stride = tp[L];

    // 1) features -> bf16
    const int nIn = NNODES * din;
    k_cvt_bf16<<<(nIn + 255) / 256, 256, 0, stream>>>(lin[L], h16, nIn);

    // 2) pack fused transposed weights + bias
    const int segBase[4] = {qb[L], kb[L], vb[L], sb[L]};
    const int segPad[4]  = {HD, HD, HD, sPad[L]};
    const int segOut[4]  = {HD, HD, HD, sOut[L]};
    for (int p = 0; p < 4; ++p) {
      const float* w = (const float*)d_in[2 + L * 8 + p * 2];
      const float* b = (const float*)d_in[2 + L * 8 + p * 2 + 1];
      const int tot = segPad[p] * din;
      k_pack_w<<<(tot + 255) / 256, 256, 0, stream>>>(w, b, Wt, biasb,
                                                      din, segOut[p], segBase[p], segPad[p]);
    }

    // 3) fused WMMA GEMM: proj = h @ [Wq|Wk|Wv|Ws] + bias  (branch-free tiles)
    if (L == 0) {
      gemm_qkvs_wmma<128, 4><<<dim3(ROWT, 4), 32, 0, stream>>>(h16, Wt, biasb, proj, 0, 256);
    } else if (L == 1) {
      gemm_qkvs_wmma<64, 4><<<dim3(ROWT, 7), 32, 0, stream>>>(h16, Wt, biasb, proj, 0, 448);
    } else {
      gemm_qkvs_wmma<64, 4><<<dim3(ROWT, 4), 32, 0, stream>>>(h16, Wt, biasb, proj, 0, 288);
      gemm_qkvs_wmma<64, 2><<<dim3(ROWT, 1), 32, 0, stream>>>(h16, Wt, biasb, proj, 256, 288);
    }

    // 4) init segment-softmax state
    const int nh = NNODES * H;
    k_fill_u32<<<(nh + 255) / 256, 256, 0, stream>>>(amax, ENC_NEG_INF, nh);
    k_fill_u32<<<(nh + 255) / 256, 256, 0, stream>>>((unsigned*)denomb, 0u, nh);
    const int na = NNODES * HD;
    k_fill_u32<<<(na + 255) / 256, 256, 0, stream>>>((unsigned*)agg, 0u, na);

    // 5) edge phase
    const int EH = NEDGES * H;
    const float scale = 1.0f / sqrtf((float)D);
    k_edge_logits<<<(EH + 255) / 256, 256, 0, stream>>>(src, dst, proj, stride,
                                                        qb[L], kb[L], D, scale,
                                                        alpha, amax, EH);
    k_edge_exp<<<(EH + 255) / 256, 256, 0, stream>>>(dst, alpha, amax, denomb, EH);
    const int totS = NEDGES * HD;
    k_edge_scatter<<<(totS + 255) / 256, 256, 0, stream>>>(src, dst, proj, stride, vb[L],
                                                           alpha, denomb, agg, D, HD, totS);

    // 6) finalize
    if (concat[L]) {
      const int tot = NNODES * HD;
      k_finalize_concat<<<(tot + 255) / 256, 256, 0, stream>>>(agg, proj, stride, sb[L],
                                                               lout[L], HD, relu[L], tot);
    } else {
      const int tot = NNODES * D;
      k_finalize_mean<<<(tot + 255) / 256, 256, 0, stream>>>(agg, proj, stride, sb[L],
                                                             lout[L], D, HD, relu[L], tot);
    }
  }

  // 7) row-wise log-softmax over 40 classes -> d_out
  k_log_softmax<<<(NNODES + 255) / 256, 256, 0, stream>>>(h1, (float*)d_out, 40, NNODES);
}
